// SpatialSemanticIdentifier_78400333021748
// MI455X (gfx1250) — compile-verified, matrix operand order checked by reading, not run on hardware
//
#include <hip/hip_runtime.h>
#include <hip/hip_bf16.h>

// ---------------------------------------------------------------------------
// Problem constants (from reference)
// ---------------------------------------------------------------------------
#define S_   3
#define B_   4096
#define DIN_ 1024
#define H1_  512
#define H2_  384
#define H3_  256
#define D_   128
#define KC_  512     // codebook size
#define C_   3       // number of codebooks
#define INV_TEMP 10.0f
#define EPS_ 1e-12f

typedef __attribute__((ext_vector_type(16))) __bf16 v16bf;
typedef __attribute__((ext_vector_type(8)))  __bf16 v8bf;
typedef __attribute__((ext_vector_type(8)))  float  v8f;

// ---------------------------------------------------------------------------
// Elementwise f32 -> bf16
// ---------------------------------------------------------------------------
__global__ void cvt_f32_bf16_kernel(const float* __restrict__ in,
                                    __bf16* __restrict__ out, long n) {
  long i = (long)blockIdx.x * blockDim.x + threadIdx.x;
  if (i < n) out[i] = (__bf16)in[i];
}

// codebooks [C,K,D] f32 -> Et [C,D,K] bf16  (transposed for GEMM B-operand)
__global__ void cvt_cb_transpose_kernel(const float* __restrict__ cb,
                                        __bf16* __restrict__ Et) {
  long i = (long)blockIdx.x * blockDim.x + threadIdx.x;
  long n = (long)C_ * KC_ * D_;
  if (i >= n) return;
  int d = (int)(i % D_);
  long r = i / D_;
  int j = (int)(r % KC_);
  int c = (int)(r / KC_);
  Et[((long)c * D_ + d) * KC_ + j] = (__bf16)cb[i];
}

// ---------------------------------------------------------------------------
// Deterministic block tree reduction (128 threads)
// ---------------------------------------------------------------------------
__device__ __forceinline__ float block_reduce_sum128(float v, float* s) {
  int t = threadIdx.x;
  s[t] = v; __syncthreads();
  #pragma unroll
  for (int st = 64; st > 0; st >>= 1) {
    if (t < st) s[t] += s[t + st];
    __syncthreads();
  }
  float r = s[0]; __syncthreads();
  return r;
}

// ee[row] = sum_d cb[row,d]^2   (row over C*K)
__global__ __launch_bounds__(128)
void ee_kernel(const float* __restrict__ cb, float* __restrict__ ee) {
  __shared__ float sh[128];
  long row = blockIdx.x;
  float v = cb[row * D_ + threadIdx.x];
  float ss = block_reduce_sum128(v * v, sh);
  if (threadIdx.x == 0) ee[row] = ss;
}

// ---------------------------------------------------------------------------
// Generic bf16 WMMA GEMM:  out[M,N] = act(A[M,K] @ W[K,N] + bias[N])
//   ACT: 0=none 1=relu 2=tanh.  Batched via grid.z with element strides.
//   Block = 256 thr (8 waves, 2x4 wave grid), block tile 64x64, K-step 32.
//   Each wave owns a 32x16 output patch -> 2 WMMAs per K-step, shared B-frag.
// ---------------------------------------------------------------------------
template <int ACT>
__global__ __launch_bounds__(256)
void gemm_bf16_kernel(const __bf16* __restrict__ A, const __bf16* __restrict__ W,
                      const float* __restrict__ bias,
                      __bf16* __restrict__ outb, float* __restrict__ outf,
                      int M, int N, int K,
                      long sA, long sW, long sBias, long sOut) {
  const int batch = blockIdx.z;
  A += (long)batch * sA;
  W += (long)batch * sW;
  if (bias) bias += (long)batch * sBias;
  if (outb) outb += (long)batch * sOut;
  if (outf) outf += (long)batch * sOut;

  const int m0 = blockIdx.y * 64;
  const int n0 = blockIdx.x * 64;
  const int t    = threadIdx.x;
  const int lane = t & 31;
  const int wid  = t >> 5;
  const int mw = wid >> 2;      // 0..1  -> 32-row half of M tile
  const int nw = wid & 3;       // 0..3  -> 16-col quarter of N tile

  __shared__ __align__(16) __bf16 As[64 * 32];   // [m][k]
  __shared__ __align__(16) __bf16 Wt[64 * 32];   // [n][k]

  const int halfk = (lane >> 4) * 8;
  const int l15   = lane & 15;
  const int nloc  = nw * 16 + l15;
  const int mbase = mw * 32 + l15;

  v8f acc0 = {}, acc1 = {};

  for (int k0 = 0; k0 < K; k0 += 32) {
    // stage A tile 64x32 (v8bf = 8 bf16 per thread)
    {
      int r = t >> 2, c = (t & 3) * 8;
      *(v8bf*)&As[r * 32 + c] = *(const v8bf*)&A[(long)(m0 + r) * K + k0 + c];
    }
    // stage W tile 32(k) x 64(n), transposed into Wt[n][k]
    {
      int kr = t >> 3, c = (t & 7) * 8;
      v8bf w8 = *(const v8bf*)&W[(long)(k0 + kr) * N + n0 + c];
      #pragma unroll
      for (int i = 0; i < 8; ++i) Wt[(c + i) * 32 + kr] = w8[i];
    }
    if (k0 + 32 < K) {  // speculative prefetch of next A tile
      __builtin_prefetch(&A[(long)(m0 + (t >> 2)) * K + k0 + 32], 0, 1);
    }
    __syncthreads();

    v16bf b, a0, a1;
    {
      v8bf b0 = *(const v8bf*)&Wt[nloc * 32 + halfk];
      v8bf b1 = *(const v8bf*)&Wt[nloc * 32 + 16 + halfk];
      v8bf x0 = *(const v8bf*)&As[mbase * 32 + halfk];
      v8bf x1 = *(const v8bf*)&As[mbase * 32 + 16 + halfk];
      v8bf y0 = *(const v8bf*)&As[(mbase + 16) * 32 + halfk];
      v8bf y1 = *(const v8bf*)&As[(mbase + 16) * 32 + 16 + halfk];
      #pragma unroll
      for (int i = 0; i < 8; ++i) {
        b[i]  = b0[i]; b[i + 8]  = b1[i];
        a0[i] = x0[i]; a0[i + 8] = x1[i];
        a1[i] = y0[i]; a1[i + 8] = y1[i];
      }
    }
    acc0 = __builtin_amdgcn_wmma_f32_16x16x32_bf16(
        false, a0, false, b, (short)0, acc0, false, false);
    acc1 = __builtin_amdgcn_wmma_f32_16x16x32_bf16(
        false, a1, false, b, (short)0, acc1, false, false);
    __syncthreads();
  }

  const int gn = n0 + nloc;
  const float bv = bias ? bias[gn] : 0.f;
  #pragma unroll
  for (int r = 0; r < 8; ++r) {
    int gm0 = m0 + mw * 32 + r + 8 * (lane >> 4);
    float v0 = acc0[r] + bv;
    float v1 = acc1[r] + bv;
    if (ACT == 1) { v0 = fmaxf(v0, 0.f); v1 = fmaxf(v1, 0.f); }
    if (ACT == 2) { v0 = tanhf(v0);      v1 = tanhf(v1); }
    long o0 = (long)gm0 * N + gn;
    long o1 = (long)(gm0 + 16) * N + gn;
    if (outb) { outb[o0] = (__bf16)v0; outb[o1] = (__bf16)v1; }
    if (outf) { outf[o0] = v0;         outf[o1] = v1; }
  }
}

// ---------------------------------------------------------------------------
// score[row] = sum_d T[row,d] * qw2[d]     (row over S*B)
// ---------------------------------------------------------------------------
__global__ __launch_bounds__(128)
void score_kernel(const __bf16* __restrict__ T, const float* __restrict__ qw2,
                  float* __restrict__ score) {
  __shared__ float sh[128];
  long row = blockIdx.x;
  float v = (float)T[row * D_ + threadIdx.x] * qw2[threadIdx.x];
  float tot = block_reduce_sum128(v, sh);
  if (threadIdx.x == 0) score[row] = tot;
}

// ---------------------------------------------------------------------------
// softmax over S per b, fuse embeddings; init res/res_bf/qsum
// ---------------------------------------------------------------------------
__global__ __launch_bounds__(256)
void fuse_kernel(const float* __restrict__ emb, const float* __restrict__ score,
                 float* __restrict__ res, __bf16* __restrict__ res_bf,
                 float* __restrict__ qsum) {
  int t = threadIdx.x;
  int b = blockIdx.x * 2 + (t >> 7);
  int d = t & 127;
  float s0 = score[0L * B_ + b];
  float s1 = score[1L * B_ + b];
  float s2 = score[2L * B_ + b];
  float m  = fmaxf(s0, fmaxf(s1, s2));
  float e0 = __expf(s0 - m), e1 = __expf(s1 - m), e2 = __expf(s2 - m);
  float inv = 1.f / (e0 + e1 + e2);
  float f = (e0 * emb[(0L * B_ + b) * D_ + d] +
             e1 * emb[(1L * B_ + b) * D_ + d] +
             e2 * emb[(2L * B_ + b) * D_ + d]) * inv;
  long o = (long)b * D_ + d;
  res[o]    = f;
  res_bf[o] = (__bf16)f;
  qsum[o]   = 0.f;
}

// ---------------------------------------------------------------------------
// Residual VQ step: argmin_j (ee[j] - 2*dot[b,j]) then update qsum/res
// ---------------------------------------------------------------------------
__global__ __launch_bounds__(128)
void vq_kernel(const float* __restrict__ dot, const float* __restrict__ ee,
               const float* __restrict__ cb,
               float* __restrict__ res, __bf16* __restrict__ res_bf,
               float* __restrict__ qsum, int k) {
  __shared__ float sv[128];
  __shared__ int   si[128];
  int b = blockIdx.x, t = threadIdx.x;

  float best = 3.0e38f; int bi = 0;
  for (int j = t; j < KC_; j += 128) {
    float v = ee[(long)k * KC_ + j] - 2.f * dot[(long)b * KC_ + j];
    if (v < best) { best = v; bi = j; }
  }
  sv[t] = best; si[t] = bi; __syncthreads();
  #pragma unroll
  for (int st = 64; st > 0; st >>= 1) {
    if (t < st) {
      if (sv[t + st] < sv[t] || (sv[t + st] == sv[t] && si[t + st] < si[t])) {
        sv[t] = sv[t + st]; si[t] = si[t + st];
      }
    }
    __syncthreads();
  }
  int jm = si[0];
  float q = cb[((long)k * KC_ + jm) * D_ + t];
  long o = (long)b * D_ + t;
  qsum[o] += q;
  float r = res[o] - q;
  res[o]    = r;
  res_bf[o] = (__bf16)r;
}

// ---------------------------------------------------------------------------
// Codebook regularizer: part[c] = | sum_k normalize(cb[c,k,:]) |^2
// ---------------------------------------------------------------------------
__global__ __launch_bounds__(128)
void cbloss_kernel(const float* __restrict__ cb, float* __restrict__ part) {
  __shared__ float sh[128];
  int c = blockIdx.x, t = threadIdx.x;
  float acc = 0.f;
  for (int kk = 0; kk < KC_; ++kk) {
    float e  = cb[((long)c * KC_ + kk) * D_ + t];
    float ss = block_reduce_sum128(e * e, sh);
    float nr = fmaxf(sqrtf(ss), EPS_);
    acc += e / nr;
  }
  float tot = block_reduce_sum128(acc * acc, sh);
  if (t == 0) part[c] = tot;
}

// row-normalize f32 [rows,128] -> bf16
__global__ __launch_bounds__(128)
void normalize_kernel(const float* __restrict__ in, __bf16* __restrict__ out) {
  __shared__ float sh[128];
  long row = blockIdx.x;
  int t = threadIdx.x;
  float v  = in[row * D_ + t];
  float ss = block_reduce_sum128(v * v, sh);
  float nr = fmaxf(sqrtf(ss), EPS_);
  out[row * D_ + t] = (__bf16)(v / nr);
}

// diag[row] = exp( dot(Za_row, Zb_row) / T )   -- O(B*D), off the hot path
__global__ __launch_bounds__(128)
void diag_kernel(const __bf16* __restrict__ Za, const __bf16* __restrict__ Zb,
                 float* __restrict__ diag) {
  __shared__ float sh[128];
  long row = blockIdx.x;
  int t = threadIdx.x;
  float v = (float)Za[row * D_ + t] * (float)Zb[row * D_ + t];
  float tot = block_reduce_sum128(v, sh);
  if (t == 0) diag[row] = __expf(tot * INV_TEMP);
}

// ---------------------------------------------------------------------------
// Fused contrastive GEMM: for 32 rows i of Za, over all B rows j of Zb:
//   rowsum[i] = sum_j exp(Za_i . Zb_j / T)
// A-fragments hoisted into registers; B-fragments (all 4 K-steps) loaded
// before the WMMA chain so loads clause and WMMAs issue back-to-back.
// No conditionals in the inner loop; deterministic serial per-row LDS reduction.
// ---------------------------------------------------------------------------
__global__ __launch_bounds__(256)
void expsum_kernel(const __bf16* __restrict__ Za, const __bf16* __restrict__ Zb,
                   float* __restrict__ rowsum) {
  const int i0   = blockIdx.x * 32;
  const int t    = threadIdx.x;
  const int lane = t & 31;
  const int wid  = t >> 5;
  const int mw = wid >> 2, nw = wid & 3;
  const int halfk = (lane >> 4) * 8;
  const int ncol  = nw * 16 + (lane & 15);     // 0..63 within j-tile

  // hoist A fragments for the whole K=128 (4 steps of 32)
  v16bf afrag[4];
  const long arow = (long)(i0 + mw * 16 + (lane & 15)) * D_;
  #pragma unroll
  for (int ks = 0; ks < 4; ++ks) {
    v8bf a0 = *(const v8bf*)&Za[arow + ks * 32 + halfk];
    v8bf a1 = *(const v8bf*)&Za[arow + ks * 32 + 16 + halfk];
    #pragma unroll
    for (int i = 0; i < 8; ++i) { afrag[ks][i] = a0[i]; afrag[ks][i + 8] = a1[i]; }
  }

  float rs[8];
  #pragma unroll
  for (int r = 0; r < 8; ++r) rs[r] = 0.f;

  for (int j0 = 0; j0 < B_; j0 += 64) {
    const long brow = (long)(j0 + ncol) * D_;
    v16bf bfr[4];
    #pragma unroll
    for (int ks = 0; ks < 4; ++ks) {
      v8bf b0 = *(const v8bf*)&Zb[brow + ks * 32 + halfk];
      v8bf b1 = *(const v8bf*)&Zb[brow + ks * 32 + 16 + halfk];
      #pragma unroll
      for (int i = 0; i < 8; ++i) { bfr[ks][i] = b0[i]; bfr[ks][i + 8] = b1[i]; }
    }
    v8f c = {};
    #pragma unroll
    for (int ks = 0; ks < 4; ++ks) {
      c = __builtin_amdgcn_wmma_f32_16x16x32_bf16(
          false, afrag[ks], false, bfr[ks], (short)0, c, false, false);
    }
    #pragma unroll
    for (int r = 0; r < 8; ++r) rs[r] += __expf(c[r] * INV_TEMP);
  }

  // deterministic reduction: each (row 0..31, col 0..63) written by one thread
  __shared__ float partial[32 * 64];
  #pragma unroll
  for (int r = 0; r < 8; ++r) {
    int row = mw * 16 + r + 8 * (lane >> 4);
    partial[row * 64 + ncol] = rs[r];
  }
  __syncthreads();
  if (t < 32) {
    float acc = 0.f;
    for (int cix = 0; cix < 64; ++cix) acc += partial[t * 64 + cix];
    rowsum[i0 + t] = acc;
  }
}

// ---------------------------------------------------------------------------
// Final outputs.  blocks 0..2: contrast_s.  block 3: codebook loss + zero.
// ---------------------------------------------------------------------------
__global__ __launch_bounds__(256)
void finalize_kernel(const float* __restrict__ R, const float* __restrict__ diagR,
                     const float* __restrict__ Bsum, const float* __restrict__ pos,
                     const float* __restrict__ part, float* __restrict__ out) {
  int t = threadIdx.x;
  if (blockIdx.x == 3) {
    if (t == 0) {
      out[0] = (part[0] + part[1] + part[2]) / (float)((long)C_ * KC_ * KC_);
      out[1] = 0.f;   // 0.0 * commit_loss
    }
    return;
  }
  int s = blockIdx.x;
  __shared__ float sh[256];
  float acc = 0.f;
  for (int b = t; b < B_; b += 256) {
    float denom = R[b] + Bsum[(long)s * B_ + b] - diagR[b];
    acc += logf(denom) - logf(pos[(long)s * B_ + b]);
  }
  sh[t] = acc; __syncthreads();
  #pragma unroll
  for (int st = 128; st > 0; st >>= 1) {
    if (t < st) sh[t] += sh[t + st];
    __syncthreads();
  }
  if (t == 0) out[2 + s] = sh[0] / (float)B_;
}

// ---------------------------------------------------------------------------
// Host-side orchestration
// ---------------------------------------------------------------------------
static inline size_t bump(size_t& off, size_t bytes) {
  off = (off + 255) & ~(size_t)255;
  size_t o = off;
  off += bytes;
  return o;
}

extern "C" void kernel_launch(void* const* d_in, const int* in_sizes, int n_in,
                              void* d_out, int out_size, void* d_ws, size_t ws_size,
                              hipStream_t stream) {
  (void)in_sizes; (void)n_in; (void)out_size; (void)ws_size;
  const float* x   = (const float*)d_in[0];
  const float* ew1 = (const float*)d_in[1];
  const float* eb1 = (const float*)d_in[2];
  const float* ew2 = (const float*)d_in[3];
  const float* eb2 = (const float*)d_in[4];
  const float* ew3 = (const float*)d_in[5];
  const float* eb3 = (const float*)d_in[6];
  const float* ew4 = (const float*)d_in[7];
  const float* eb4 = (const float*)d_in[8];
  const float* qw1 = (const float*)d_in[9];
  const float* qb1 = (const float*)d_in[10];
  const float* qw2 = (const float*)d_in[11];
  const float* cb  = (const float*)d_in[12];
  float* out = (float*)d_out;
  char* ws = (char*)d_ws;

  // ---- workspace layout (bump allocator; scratch overlays dead x_bf) ----
  size_t off = 0;
  __bf16* w1b  = (__bf16*)(ws + bump(off, (size_t)S_ * DIN_ * H1_ * 2));
  __bf16* w2b  = (__bf16*)(ws + bump(off, (size_t)S_ * H1_ * H2_ * 2));
  __bf16* w3b  = (__bf16*)(ws + bump(off, (size_t)S_ * H2_ * H3_ * 2));
  __bf16* w4b  = (__bf16*)(ws + bump(off, (size_t)S_ * H3_ * D_ * 2));
  __bf16* qw1b = (__bf16*)(ws + bump(off, (size_t)D_ * D_ * 2));
  __bf16* Etb  = (__bf16*)(ws + bump(off, (size_t)C_ * D_ * KC_ * 2));
  float*  eeb  = (float*) (ws + bump(off, (size_t)C_ * KC_ * 4));
  __bf16* h1b  = (__bf16*)(ws + bump(off, (size_t)S_ * B_ * H1_ * 2));
  __bf16* h2b  = (__bf16*)(ws + bump(off, (size_t)S_ * B_ * H2_ * 2));
  __bf16* h3b  = (__bf16*)(ws + bump(off, (size_t)S_ * B_ * H3_ * 2));
  __bf16* embb = (__bf16*)(ws + bump(off, (size_t)S_ * B_ * D_ * 2));
  float*  embf = (float*) (ws + bump(off, (size_t)S_ * B_ * D_ * 4));
  float*  scor = (float*) (ws + bump(off, (size_t)S_ * B_ * 4));
  float*  resf = (float*) (ws + bump(off, (size_t)B_ * D_ * 4));
  __bf16* resb = (__bf16*)(ws + bump(off, (size_t)B_ * D_ * 2));
  float*  qsum = (float*) (ws + bump(off, (size_t)B_ * D_ * 4));
  float*  part = (float*) (ws + bump(off, 4 * 4));
  size_t mark = off;
  __bf16* xb = (__bf16*)(ws + bump(off, (size_t)S_ * B_ * DIN_ * 2)); // encoder stage only
  off = mark;                                                          // overlay:
  __bf16* Tb   = (__bf16*)(ws + bump(off, (size_t)S_ * B_ * D_ * 2));
  float*  dotb = (float*) (ws + bump(off, (size_t)B_ * KC_ * 4));
  __bf16* z1b  = (__bf16*)(ws + bump(off, (size_t)B_ * D_ * 2));
  __bf16* z2b  = (__bf16*)(ws + bump(off, (size_t)S_ * B_ * D_ * 2));
  float*  Rb   = (float*) (ws + bump(off, (size_t)B_ * 4));
  float*  dRb  = (float*) (ws + bump(off, (size_t)B_ * 4));
  float*  Bsum = (float*) (ws + bump(off, (size_t)S_ * B_ * 4));
  float*  posb = (float*) (ws + bump(off, (size_t)S_ * B_ * 4));

  auto cvt = [&](const float* src, __bf16* dst, long n) {
    int blocks = (int)((n + 255) / 256);
    cvt_f32_bf16_kernel<<<blocks, 256, 0, stream>>>(src, dst, n);
  };

  // ---- stage 0: precision conversion ----
  cvt(x,   xb,   (long)S_ * B_ * DIN_);
  cvt(ew1, w1b,  (long)S_ * DIN_ * H1_);
  cvt(ew2, w2b,  (long)S_ * H1_ * H2_);
  cvt(ew3, w3b,  (long)S_ * H2_ * H3_);
  cvt(ew4, w4b,  (long)S_ * H3_ * D_);
  cvt(qw1, qw1b, (long)D_ * D_);
  {
    long n = (long)C_ * KC_ * D_;
    cvt_cb_transpose_kernel<<<(int)((n + 255) / 256), 256, 0, stream>>>(cb, Etb);
  }
  ee_kernel<<<C_ * KC_, 128, 0, stream>>>(cb, eeb);

  // ---- stage 1: per-source MLP (batched WMMA GEMMs) ----
  gemm_bf16_kernel<1><<<dim3(H1_ / 64, B_ / 64, S_), 256, 0, stream>>>(
      xb, w1b, eb1, h1b, nullptr, B_, H1_, DIN_,
      (long)B_ * DIN_, (long)DIN_ * H1_, H1_, (long)B_ * H1_);
  gemm_bf16_kernel<1><<<dim3(H2_ / 64, B_ / 64, S_), 256, 0, stream>>>(
      h1b, w2b, eb2, h2b, nullptr, B_, H2_, H1_,
      (long)B_ * H1_, (long)H1_ * H2_, H2_, (long)B_ * H2_);
  gemm_bf16_kernel<1><<<dim3(H3_ / 64, B_ / 64, S_), 256, 0, stream>>>(
      h2b, w3b, eb3, h3b, nullptr, B_, H3_, H2_,
      (long)B_ * H2_, (long)H2_ * H3_, H3_, (long)B_ * H3_);
  gemm_bf16_kernel<0><<<dim3(D_ / 64, B_ / 64, S_), 256, 0, stream>>>(
      h3b, w4b, eb4, embb, embf, B_, D_, H3_,
      (long)B_ * H3_, (long)H3_ * D_, D_, (long)B_ * D_);

  // ---- stage 2: attention fusion ----
  gemm_bf16_kernel<2><<<dim3(D_ / 64, (S_ * B_) / 64, 1), 256, 0, stream>>>(
      embb, qw1b, qb1, Tb, nullptr, S_ * B_, D_, D_, 0, 0, 0, 0);
  score_kernel<<<S_ * B_, 128, 0, stream>>>(Tb, qw2, scor);
  fuse_kernel<<<B_ / 2, 256, 0, stream>>>(embf, scor, resf, resb, qsum);

  // ---- stage 3: residual VQ (3 codebooks) ----
  for (int c = 0; c < C_; ++c) {
    gemm_bf16_kernel<0><<<dim3(KC_ / 64, B_ / 64, 1), 256, 0, stream>>>(
        resb, Etb + (long)c * D_ * KC_, nullptr, nullptr, dotb,
        B_, KC_, D_, 0, 0, 0, 0);
    vq_kernel<<<B_, 128, 0, stream>>>(dotb, eeb, cb, resf, resb, qsum, c);
  }

  // ---- stage 4: codebook regularizer ----
  cbloss_kernel<<<C_, 128, 0, stream>>>(cb, part);

  // ---- stage 5: contrastive ----
  normalize_kernel<<<B_, 128, 0, stream>>>(qsum, z1b);
  normalize_kernel<<<S_ * B_, 128, 0, stream>>>(embf, z2b);
  diag_kernel<<<B_, 128, 0, stream>>>(z1b, z1b, dRb);
  expsum_kernel<<<B_ / 32, 256, 0, stream>>>(z1b, z1b, Rb);
  for (int s = 0; s < S_; ++s) {
    const __bf16* z2s = z2b + (long)s * B_ * D_;
    diag_kernel<<<B_, 128, 0, stream>>>(z1b, z2s, posb + (long)s * B_);
    expsum_kernel<<<B_ / 32, 256, 0, stream>>>(z1b, z2s, Bsum + (long)s * B_);
  }

  // ---- stage 6: outputs ----
  finalize_kernel<<<4, 256, 0, stream>>>(Rb, dRb, Bsum, posb, part, out);
}
